// MultiHeadSelfAttention_86560770883815
// MI455X (gfx1250) — matrix operand order repaired and checked by reading, not compiled, not run on hardware
//
#include <hip/hip_runtime.h>
#include <math.h>

typedef __attribute__((ext_vector_type(16))) __bf16 v16bf;
typedef __attribute__((ext_vector_type(8)))  float  v8f;

#define B_ 2
#define T_ 2048
#define C_ 1024
#define H_ 16
#define D_ 64

// ---------------------------------------------------------------- helpers
__device__ __forceinline__ unsigned short f2bf(float f) {
    union { float f; unsigned int u; } cv; cv.f = f;
    unsigned int u = cv.u;
    unsigned int r = u + 0x7FFFu + ((u >> 16) & 1u);   // round-to-nearest-even
    return (unsigned short)(r >> 16);
}

// CDNA5 async global->LDS copy (ASYNCcnt-tracked), 16 bytes per lane.
// lds_off: per-lane LDS byte address; gptr: per-lane global address.
__device__ __forceinline__ void async_copy_b128(unsigned lds_off, const void* gptr) {
    unsigned long long ga = (unsigned long long)gptr;
    asm volatile("global_load_async_to_lds_b128 %0, %1, off"
                 :: "v"(lds_off), "v"(ga) : "memory");
}
__device__ __forceinline__ void wait_async0() {
    asm volatile("s_wait_asynccnt 0x0" ::: "memory");
}
__device__ __forceinline__ unsigned lds_addr(const void* p) {
    return (unsigned)(unsigned long long)p;   // low 32 bits = wave-relative LDS address
}

__global__ void cvt_f32_to_bf16(const float* __restrict__ src,
                                unsigned short* __restrict__ dst, int n) {
    int i = blockIdx.x * blockDim.x + threadIdx.x;
    if (i < n) dst[i] = f2bf(src[i]);
}

// ---------------------------------------------------------------- generic WMMA GEMM
// C[M,N] = scale * (A[M,K] @ Bcm[N,K]^T) + bias[N]
//   A: bf16 row-major (or f32 row-major when A_F32, converted while staging)
//   Bcm: bf16, laid out [N][K] ("column-major" B, row stride K)
// Batched over blockIdx.z with element strides aBatch/bBatch;
// C base = (z/cDiv)*cOuter + (z%cDiv)*cInner, row stride ldc.
// NT_C: stream C with non-temporal stores (output won't fit in L2).
template<bool A_F32, bool OUT_BF16, bool NT_C>
__global__ __launch_bounds__(256)
void gemm_bf16_wmma(const void* __restrict__ Aptr,
                    const unsigned short* __restrict__ Bcm,
                    void* __restrict__ Cptr,
                    const float* __restrict__ bias,
                    float scale, int M, int N, int K,
                    long long aBatch, long long bBatch,
                    int cDiv, long long cOuter, long long cInner, int ldc)
{
    __shared__ unsigned short As[2][128 * 32];  // double-buffered 128 M-rows x 32 K (bf16)
    __shared__ unsigned short Bs[2][64 * 32];   // double-buffered 64 N-rows  x 32 K (bf16)

    const int tid  = threadIdx.x;
    const int lane = tid & 31;
    const int wave = tid >> 5;
    const int wm   = wave & 3;     // wave's 32-row group within 128
    const int wn   = wave >> 2;    // wave's 32-col group within 64
    const int r    = lane & 15;
    const int h    = lane >> 4;

    const int z  = blockIdx.z;
    const int m0 = blockIdx.y * 128;
    const int n0 = blockIdx.x * 64;

    const size_t aBase = (size_t)z * (size_t)aBatch;
    const size_t bBase = (size_t)z * (size_t)bBatch;
    const size_t cBase = (size_t)(z / cDiv) * (size_t)cOuter
                       + (size_t)(z % cDiv) * (size_t)cInner;

    const int arow = tid >> 1, aseg = tid & 1;  // A stage: 128 rows x 2 segs x 16 halves
    const int brow = tid >> 2, bseg = tid & 3;  // B stage: 64 rows x 4 segs x 8 halves

    // Issue the stage for K-chunk kt into LDS buffer bb.
    auto stageTile = [&](int kt, int bb) {
        if (A_F32) {
            // f32 attention matrix: synchronous load (non-temporal) + convert + LDS store
            const float* srcf = (const float*)Aptr + aBase
                + (size_t)(m0 + arow) * (size_t)K + (size_t)(kt * 32 + aseg * 16);
            unsigned short* dst = &As[bb][arow * 32 + aseg * 16];
            #pragma unroll
            for (int q = 0; q < 16; ++q) dst[q] = f2bf(__builtin_nontemporal_load(srcf + q));
        } else {
            const unsigned short* srch = (const unsigned short*)Aptr + aBase
                + (size_t)(m0 + arow) * (size_t)K + (size_t)(kt * 32 + aseg * 16);
            async_copy_b128(lds_addr(&As[bb][arow * 32 + aseg * 16]),     srch);
            async_copy_b128(lds_addr(&As[bb][arow * 32 + aseg * 16 + 8]), srch + 8);
        }
        const unsigned short* srcb = Bcm + bBase
            + (size_t)(n0 + brow) * (size_t)K + (size_t)(kt * 32 + bseg * 8);
        async_copy_b128(lds_addr(&Bs[bb][brow * 32 + bseg * 8]), srcb);
    };

    v8f acc[2][2];
    #pragma unroll
    for (int i = 0; i < 2; ++i)
        #pragma unroll
        for (int j = 0; j < 2; ++j)
            #pragma unroll
            for (int q = 0; q < 8; ++q) acc[i][j][q] = 0.0f;

    const int nk = K >> 5;
    stageTile(0, 0);

    for (int kt = 0; kt < nk; ++kt) {
        const int cur = kt & 1;
        // own async writes into buf[cur] done; barrier => all waves' writes visible
        // and all waves finished reading buf[cur^1] in the previous iteration.
        wait_async0();
        __syncthreads();
        if (kt + 1 < nk) stageTile(kt + 1, cur ^ 1);   // overlaps with WMMAs below

        // ---- per-lane fragment loads per ISA 7.12.2 layouts ----
        union FragU { uint4 u[2]; v16bf v; };
        FragU fa[2], fb[2];
        #pragma unroll
        for (int i = 0; i < 2; ++i) {
            int mb = (wm * 32 + i * 16 + r) * 32;
            fa[i].u[0] = *reinterpret_cast<const uint4*>(&As[cur][mb + h * 8]);       // K = h*8..h*8+7
            fa[i].u[1] = *reinterpret_cast<const uint4*>(&As[cur][mb + 16 + h * 8]);  // K = 16+h*8..
        }
        #pragma unroll
        for (int j = 0; j < 2; ++j) {
            int nb = (wn * 32 + j * 16 + r) * 32;
            fb[j].u[0] = *reinterpret_cast<const uint4*>(&Bs[cur][nb + h * 8]);       // K = h*8..h*8+7
            fb[j].u[1] = *reinterpret_cast<const uint4*>(&Bs[cur][nb + 16 + h * 8]);  // K = 16+h*8..
        }
        #pragma unroll
        for (int i = 0; i < 2; ++i)
            #pragma unroll
            for (int j = 0; j < 2; ++j)
                acc[i][j] = __builtin_amdgcn_wmma_f32_16x16x32_bf16(
                    false, fa[i].v, false, fb[j].v, (short)0, acc[i][j], false, false);
    }

    // ---- epilogue: C/D layout M = q + 8*h, N = r ----
    #pragma unroll
    for (int i = 0; i < 2; ++i) {
        #pragma unroll
        for (int j = 0; j < 2; ++j) {
            int mt = m0 + wm * 32 + i * 16 + 8 * h;
            int nt = n0 + wn * 32 + j * 16 + r;
            float bv = bias ? bias[nt] : 0.0f;
            #pragma unroll
            for (int q = 0; q < 8; ++q) {
                float val = acc[i][j][q] * scale + bv;
                size_t offc = cBase + (size_t)(mt + q) * (size_t)ldc + (size_t)nt;
                if (OUT_BF16) {
                    ((unsigned short*)Cptr)[offc] = f2bf(val);
                } else if (NT_C) {
                    __builtin_nontemporal_store(val, (float*)Cptr + offc);
                } else {
                    ((float*)Cptr)[offc] = val;
                }
            }
        }
    }
}

// ---------------------------------------------------------------- split qkv -> Qb, Kb, Vt(bf16)
__global__ void split_qkv(const float* __restrict__ qkv,
                          unsigned short* __restrict__ Qb,
                          unsigned short* __restrict__ Kb,
                          unsigned short* __restrict__ Vtb)
{
    int idx = blockIdx.x * 256 + threadIdx.x;       // over B*T*C
    int c  = idx & (C_ - 1);
    int bt = idx >> 10;                              // C_ = 1024
    int t  = bt & (T_ - 1);
    int b  = bt >> 11;                               // T_ = 2048
    int hh = c >> 6;
    int d  = c & 63;
    size_t rowb = (size_t)bt * (size_t)(3 * C_);
    float qv = qkv[rowb + c];
    float kv = qkv[rowb + C_ + c];
    float vv = qkv[rowb + 2 * C_ + c];
    size_t bh = (size_t)(b * H_ + hh);
    Qb [(bh * T_ + t) * D_ + d] = f2bf(qv);
    Kb [(bh * T_ + t) * D_ + d] = f2bf(kv);
    Vtb[(bh * D_ + d) * T_ + t] = f2bf(vv);          // V stored transposed [b,h,d,t]
}

// ---------------------------------------------------------------- fused softmax + 5-tap Gaussian (in-place per row)
__global__ __launch_bounds__(256)
void softmax_smooth_rows(float* __restrict__ attn)
{
    __shared__ float buf[T_];
    __shared__ float red[256];
    const int tid = threadIdx.x;
    float* row = attn + (size_t)blockIdx.x * (size_t)T_;

    float m = -3.4e38f;
    #pragma unroll
    for (int i = 0; i < T_ / 256; ++i) {
        float v = __builtin_nontemporal_load(row + tid + i * 256);
        buf[tid + i * 256] = v;
        m = fmaxf(m, v);
    }
    red[tid] = m;
    __syncthreads();
    for (int s = 128; s > 0; s >>= 1) {
        if (tid < s) red[tid] = fmaxf(red[tid], red[tid + s]);
        __syncthreads();
    }
    const float mx = red[0];
    __syncthreads();

    float ssum = 0.0f;
    #pragma unroll
    for (int i = 0; i < T_ / 256; ++i) {
        int idx = tid + i * 256;
        float e = __expf(buf[idx] - mx);
        buf[idx] = e;
        ssum += e;
    }
    red[tid] = ssum;
    __syncthreads();
    for (int s = 128; s > 0; s >>= 1) {
        if (tid < s) red[tid] += red[tid + s];
        __syncthreads();
    }
    const float inv = 1.0f / red[0];
    __syncthreads();

    #pragma unroll
    for (int i = 0; i < T_ / 256; ++i) buf[tid + i * 256] *= inv;
    __syncthreads();

    // Gaussian kernel: exp(-{4,1,0,1,4}/2), normalized; zero-padded 'same' conv
    const float w0e  = 0.13533528323661270f;   // exp(-2)
    const float w1e  = 0.60653065971263342f;   // exp(-0.5)
    const float invs = 1.0f / (1.0f + 2.0f * w1e + 2.0f * w0e);
    const float c0 = w0e * invs, c1 = w1e * invs, c2 = invs;

    #pragma unroll
    for (int i = 0; i < T_ / 256; ++i) {
        int idx = tid + i * 256;
        float l2 = (idx >= 2)      ? buf[idx - 2] : 0.0f;
        float l1 = (idx >= 1)      ? buf[idx - 1] : 0.0f;
        float cc = buf[idx];
        float r1 = (idx + 1 < T_)  ? buf[idx + 1] : 0.0f;
        float r2 = (idx + 2 < T_)  ? buf[idx + 2] : 0.0f;
        float o  = c0 * (l2 + r2) + c1 * (l1 + r1) + c2 * cc;
        __builtin_nontemporal_store(o, row + idx);
    }
}

// ---------------------------------------------------------------- launch
extern "C" void kernel_launch(void* const* d_in, const int* in_sizes, int n_in,
                              void* d_out, int out_size, void* d_ws, size_t ws_size,
                              hipStream_t stream)
{
    const float* x    = (const float*)d_in[0];
    const float* Wqkv = (const float*)d_in[1];
    const float* bqkv = (const float*)d_in[2];
    const float* Wout = (const float*)d_in[3];
    const float* bout = (const float*)d_in[4];

    float* out  = (float*)d_out;                                   // [B,T,C]
    float* attn = (float*)d_out + (size_t)B_ * T_ * C_;            // [B,H,T,T]

    char* ws = (char*)d_ws;
    size_t off = 0;
    auto alloc = [&](size_t bytes) -> void* {
        void* p = ws + off;
        off += (bytes + 255) & ~(size_t)255;
        return p;
    };
    unsigned short* xb    = (unsigned short*)alloc((size_t)B_ * T_ * C_ * 2);
    unsigned short* Wqkvb = (unsigned short*)alloc((size_t)3 * C_ * C_ * 2);
    unsigned short* Woutb = (unsigned short*)alloc((size_t)C_ * C_ * 2);
    float*          qkvf  = (float*)         alloc((size_t)B_ * T_ * 3 * C_ * 4);
    unsigned short* Qb    = (unsigned short*)alloc((size_t)B_ * H_ * T_ * D_ * 2);
    unsigned short* Kb    = (unsigned short*)alloc((size_t)B_ * H_ * T_ * D_ * 2);
    unsigned short* Vtb   = (unsigned short*)alloc((size_t)B_ * H_ * T_ * D_ * 2);
    unsigned short* ctxb  = (unsigned short*)alloc((size_t)B_ * T_ * C_ * 2);

    // 1) fp32 -> bf16 conversions
    { int n = B_ * T_ * C_;   cvt_f32_to_bf16<<<n / 256, 256, 0, stream>>>(x, xb, n); }
    { int n = 3 * C_ * C_;    cvt_f32_to_bf16<<<n / 256, 256, 0, stream>>>(Wqkv, Wqkvb, n); }
    { int n = C_ * C_;        cvt_f32_to_bf16<<<n / 256, 256, 0, stream>>>(Wout, Woutb, n); }

    // 2) qkv = x @ Wqkv^T + bqkv   (M=4096, N=3072, K=1024)
    gemm_bf16_wmma<false, false, false><<<dim3(3 * C_ / 64, B_ * T_ / 128, 1), 256, 0, stream>>>(
        xb, Wqkvb, qkvf, bqkv, 1.0f,
        B_ * T_, 3 * C_, C_,
        0LL, 0LL, 1, 0LL, 0LL, 3 * C_);

    // 3) split into Q, K (bf16 [b,h,t,d]) and V transposed (bf16 [b,h,d,t])
    split_qkv<<<(B_ * T_ * C_) / 256, 256, 0, stream>>>(qkvf, Qb, Kb, Vtb);

    // 4) scores = (Q @ K^T) / sqrt(D)  -> d_out attn region, streamed NT
    gemm_bf16_wmma<false, false, true><<<dim3(T_ / 64, T_ / 128, B_ * H_), 256, 0, stream>>>(
        Qb, Kb, attn, nullptr, 0.125f,
        T_, T_, D_,
        (long long)T_ * D_, (long long)T_ * D_,
        1, (long long)T_ * T_, 0LL, T_);

    // 5) fused softmax + Gaussian smoothing, in place (one NT RMW pass over the 536MB matrix)
    softmax_smooth_rows<<<B_ * H_ * T_, 256, 0, stream>>>(attn);

    // 6) ctx = attn @ V   (A is f32 smoothed attn, NT loads + cvt on the fly)
    gemm_bf16_wmma<true, true, false><<<dim3(D_ / 64, T_ / 128, B_ * H_), 256, 0, stream>>>(
        attn, Vtb, ctxb, nullptr, 1.0f,
        T_, D_, T_,
        (long long)T_ * T_, (long long)D_ * T_,
        H_, (long long)T_ * C_, (long long)D_, C_);

    // 7) out = ctx @ Wout^T + bout  (M=4096, N=1024, K=1024)
    gemm_bf16_wmma<false, false, false><<<dim3(C_ / 64, B_ * T_ / 128, 1), 256, 0, stream>>>(
        ctxb, Woutb, out, bout, 1.0f,
        B_ * T_, C_, C_,
        0LL, 0LL, 1, 0LL, 0LL, C_);
}
